// Pnet_63221918597222
// MI455X (gfx1250) — matrix operand order repaired and checked by reading, not compiled, not run on hardware
//
#include <hip/hip_runtime.h>
#include <hip/hip_bf16.h>

// MI455X (gfx1250) implementation.
// Phases:
//  K0: zero h ping-pong buffers + grid-barrier state (ws is not re-poisoned between replays)
//  K1: b_comb = b_lgssm @ W_ih + b_ih                              (tiny VALU)
//  K2: W_comb^T(bf16, K-major) = (W_lgssm @ W_ih)^T                (WMMA bf16, 51.5 GMAC)
//  K3: GI(f32)      = Z @ W_comb + b_comb                          (WMMA bf16, 51.5 GMAC,
//      B tile staged LDS via global_load_async_to_lds_b128 / ASYNCcnt)
//  K4: persistent 64-WG GRU recurrence, 512 steps, W_hh slice in LDS (bf16),
//      device-wide sense barrier per step; writes H_all (bf16)
//  K5: Dpad(bf16, 128 cols zero-padded) = H_all @ W_dense + b_dense (WMMA, async A stage)
//  K6: out = mu + softplus(sigma) * noise, mu/sigma via WMMA heads (K padded 100->128)

typedef __bf16 bf16;
typedef __attribute__((ext_vector_type(16))) __bf16   v16bf;
typedef __attribute__((ext_vector_type(8)))  float    v8f;
typedef __attribute__((ext_vector_type(8)))  unsigned v8u;

static __device__ __forceinline__ v8f zero8() {
  return (v8f){0.f, 0.f, 0.f, 0.f, 0.f, 0.f, 0.f, 0.f};
}

// Async 16B global -> LDS copy (GLOBAL_LOAD_ASYNC_TO_LDS_B128, tracked by ASYNCcnt).
// LDS operand = low 32 bits of the generic pointer (== LDS byte offset).
static __device__ __forceinline__ void async_copy_b128(const void* gaddr, void* lds) {
  const unsigned l = (unsigned)(size_t)lds;
  asm volatile("global_load_async_to_lds_b128 %0, %1, off"
               :
               : "v"(l), "v"(gaddr)
               : "memory");
}
static __device__ __forceinline__ void wait_async0() {
  asm volatile("s_wait_asynccnt 0x0" ::: "memory");
}

// A-fragment (16x32 bf16, row-major, contiguous K run of >=32 at `row`).
// ISA 7.12.2: lanes 0-15 -> K {0..7,16..23}, lanes 16-31 -> K {8..15,24..31}.
static __device__ __forceinline__ v16bf frag_a(const bf16* row, int lane) {
  const int khalf = (lane >> 4) << 3; // 0 or 8
  v8u u;
#pragma unroll
  for (int r = 0; r < 8; ++r) {
    const int kb = ((r & 4) << 2) + khalf + ((r & 3) << 1);
    u[r] = *reinterpret_cast<const unsigned*>(row + kb);
  }
  return __builtin_bit_cast(v16bf, u);
}

// B-fragment (32x16 bf16). `col` points at this column's contiguous K-major run.
// VGPR r holds K = 16*(r>=4) + 8*(lane>=16) + 2(r&3), +1 for column n = lane&15.
static __device__ __forceinline__ v16bf frag_b(const bf16* col, int lane) {
  const int kh = (lane >> 4) << 3; // 0 or 8
  v8u u;
#pragma unroll
  for (int r = 0; r < 8; ++r) {
    u[r] = *reinterpret_cast<const unsigned*>(col + ((r & 4) << 2) + kh + ((r & 3) << 1));
  }
  return __builtin_bit_cast(v16bf, u);
}

static __device__ __forceinline__ v8f wmma_bf16(v16bf a, v16bf b, v8f c) {
  return __builtin_amdgcn_wmma_f32_16x16x32_bf16(false, a, false, b, (short)0, c,
                                                 false, false);
}

static __device__ __forceinline__ float sigmoidf_(float x) {
  return 1.f / (1.f + __expf(-x));
}

// ---------------------------------------------------------------------------
// Generic 128x128-tile WMMA GEMM: C(MxN) = A(MxK) @ B(KxN) [+ bias[n]]
// TA/TB in {float, bf16}; TC in {float, bf16}. M, K multiples of 128/32.
// ZPAD  : write zeros for n in [N, n0+128) (zero-pads the dense output).
// BKMAJ : B is stored K-major ([n][k], ldb = k-stride); staged via async DMA.
// TRC   : store C transposed (C[n*ldc + m], bf16; vectorized 16B lane stores).
// bf16 A is staged via async DMA ([m][k] is already a straight 2D block copy).
// ---------------------------------------------------------------------------
template <typename TA, typename TB, typename TC, bool ZPAD, bool BKMAJ, bool TRC>
__global__ __launch_bounds__(256) void gemm_wmma(
    const TA* __restrict__ A, int lda, const TB* __restrict__ B, int ldb,
    TC* __restrict__ C, int ldc, const float* __restrict__ bias, int N, int K) {
  __shared__ __align__(16) bf16 As[128 * 32]; // [m][k]
  __shared__ __align__(16) bf16 Bs[128 * 32]; // [n][k] (transposed stage)
  const int tid = threadIdx.x;
  const int lane = tid & 31;
  const int wave = tid >> 5;
  const int m0 = blockIdx.y << 7;
  const int n0 = blockIdx.x << 7;
  const int wm = (wave & 3) << 5;  // 4 waves along M (32 rows each)
  const int wn = (wave >> 2) << 6; // 2 waves along N (64 cols each)
  const bool nedge = (n0 + 128 > N);

  v8f acc[2][4];
#pragma unroll
  for (int i = 0; i < 2; ++i)
#pragma unroll
    for (int j = 0; j < 4; ++j) acc[i][j] = zero8();

  for (int kc = 0; kc < K; kc += 32) {
    if (kc + 64 < K) { // prefetch next-next k-chunk -> global_prefetch_b8
      __builtin_prefetch(A + (size_t)(m0 + (tid >> 1)) * lda + kc + 64, 0, 0);
      if constexpr (!BKMAJ)
        __builtin_prefetch(B + (size_t)(kc + 64 + (tid >> 3)) * ldb + n0, 0, 0);
    }
    // ---- stage A tile (128x32) ----
    if constexpr (sizeof(TA) == 2) {
      // straight 2D block copy: 512 x 16B chunks, async global -> LDS
#pragma unroll
      for (int i = 0; i < 2; ++i) {
        const int c = tid + (i << 8); // 0..511
        const int m = c >> 2;
        const int k8 = (c & 3) << 3;
        async_copy_b128(A + (size_t)(m0 + m) * lda + kc + k8, &As[m * 32 + k8]);
      }
    } else {
#pragma unroll
      for (int i = 0; i < 4; ++i) {
        const int c = tid + (i << 8);
        const int m = c >> 3;
        const int k4 = (c & 7) << 2;
        const float4 v =
            *reinterpret_cast<const float4*>(A + (size_t)(m0 + m) * lda + kc + k4);
        bf16* lp = &As[m * 32 + k4];
        lp[0] = (bf16)v.x; lp[1] = (bf16)v.y;
        lp[2] = (bf16)v.z; lp[3] = (bf16)v.w;
      }
    }
    // ---- stage B tile ----
    if constexpr (BKMAJ) {
      // B already [n][k]: straight 2D block copy via async DMA
      if (!nedge) {
#pragma unroll
        for (int i = 0; i < 2; ++i) {
          const int c = tid + (i << 8); // 0..511
          const int n = c >> 2;
          const int k8 = (c & 3) << 3;
          async_copy_b128(B + (size_t)(n0 + n) * ldb + kc + k8, &Bs[n * 32 + k8]);
        }
      } else {
#pragma unroll
        for (int i = 0; i < 4; ++i) {
          const int c = tid + (i << 8);
          const int n = c >> 3;
          const int k4 = (c & 7) << 2;
#pragma unroll
          for (int j = 0; j < 4; ++j)
            Bs[n * 32 + k4 + j] = (n0 + n < N)
                                      ? B[(size_t)(n0 + n) * ldb + kc + k4 + j]
                                      : (bf16)0.f;
        }
      }
    } else {
      // B row-major [k][n]: transpose while staging (with fp32->bf16 convert)
#pragma unroll
      for (int i = 0; i < 4; ++i) {
        const int c = tid + (i << 8);
        const int k = c >> 5;
        const int n4 = (c & 31) << 2;
        const TB* gp = B + (size_t)(kc + k) * ldb + (n0 + n4);
        if (!nedge) {
          if constexpr (sizeof(TB) == 4) {
            const float4 v = *reinterpret_cast<const float4*>(gp);
            Bs[(n4 + 0) * 32 + k] = (bf16)v.x;
            Bs[(n4 + 1) * 32 + k] = (bf16)v.y;
            Bs[(n4 + 2) * 32 + k] = (bf16)v.z;
            Bs[(n4 + 3) * 32 + k] = (bf16)v.w;
          } else {
            const uint2 v = *reinterpret_cast<const uint2*>(gp);
            union { unsigned u; bf16 h[2]; } p0, p1;
            p0.u = v.x; p1.u = v.y;
            Bs[(n4 + 0) * 32 + k] = p0.h[0];
            Bs[(n4 + 1) * 32 + k] = p0.h[1];
            Bs[(n4 + 2) * 32 + k] = p1.h[0];
            Bs[(n4 + 3) * 32 + k] = p1.h[1];
          }
        } else {
#pragma unroll
          for (int j = 0; j < 4; ++j) {
            const int n = n0 + n4 + j;
            Bs[(n4 + j) * 32 + k] = (n < N) ? (bf16)gp[j] : (bf16)0.f;
          }
        }
      }
    }
    if constexpr (sizeof(TA) == 2 || BKMAJ) wait_async0();
    __syncthreads();
    // ---- compute: 2x4 C tiles per wave ----
    v16bf af[2], bfr[4];
#pragma unroll
    for (int i = 0; i < 2; ++i)
      af[i] = frag_a(&As[(wm + (i << 4) + (lane & 15)) * 32], lane);
#pragma unroll
    for (int j = 0; j < 4; ++j)
      bfr[j] = frag_b(&Bs[(wn + (j << 4) + (lane & 15)) * 32], lane);
#pragma unroll
    for (int i = 0; i < 2; ++i)
#pragma unroll
      for (int j = 0; j < 4; ++j) acc[i][j] = wmma_bf16(af[i], bfr[j], acc[i][j]);
    __syncthreads();
  }

  const int crow = (lane >> 4) << 3;
  const int ccol = lane & 15;
  if constexpr (TRC) {
    // Transposed store: lane owns column n; rows m..m+7 are contiguous -> 16B store.
#pragma unroll
    for (int i = 0; i < 2; ++i)
#pragma unroll
      for (int j = 0; j < 4; ++j) {
        const size_t n = (size_t)n0 + wn + (j << 4) + ccol;
        const size_t m = (size_t)m0 + wm + (i << 4) + crow;
        alignas(16) bf16 h8[8];
#pragma unroll
        for (int r = 0; r < 8; ++r) h8[r] = (bf16)acc[i][j][r];
        *reinterpret_cast<uint4*>(&C[n * ldc + m]) =
            *reinterpret_cast<const uint4*>(h8);
      }
  } else {
#pragma unroll
    for (int i = 0; i < 2; ++i)
#pragma unroll
      for (int j = 0; j < 4; ++j)
#pragma unroll
        for (int r = 0; r < 8; ++r) {
          const size_t m = (size_t)m0 + wm + (i << 4) + crow + r;
          const int n = n0 + wn + (j << 4) + ccol;
          if (n < N) {
            float v = acc[i][j][r];
            if (bias) v += bias[n];
            C[m * ldc + n] = (TC)v;
          } else if (ZPAD) {
            C[m * ldc + n] = (TC)0.f;
          }
        }
  }
}

// ---------------------------------------------------------------------------
// K4: persistent GRU recurrence. 64 WGs; WG owns 16 hidden cols per gate.
// ---------------------------------------------------------------------------
__global__ __launch_bounds__(256) void gru_recur(
    const float* __restrict__ Whh, // (1024, 3072)
    const float* __restrict__ bhh, // (3072)
    const float* __restrict__ GI,  // (4096, 3072), row = b*512 + t
    bf16* __restrict__ hbuf,       // 2 x (16 x 1024), rows 8..15 stay zero
    bf16* __restrict__ Hall,       // (4096, 1024),  row = b*512 + t
    unsigned* __restrict__ bar) {  // [0]=count, [1]=generation
  extern __shared__ __align__(16) char smem[];
  bf16* whh_s = reinterpret_cast<bf16*>(smem); // [3][16][1024]: (g*16+j)*1024+k
  float* acc = reinterpret_cast<float*>(smem + 3 * 16 * 1024 * sizeof(bf16)); // [3][16][16]
  float* hown = acc + 3 * 256;   // [8][16] fp32 h slice
  float* bias_s = hown + 128;    // [48]
  const int tid = threadIdx.x;
  const int lane = tid & 31;
  const int wave = tid >> 5;
  const int col0 = (int)blockIdx.x << 4;

  for (int idx = tid; idx < 3 * 16 * 1024; idx += 256) {
    const int g = idx >> 14;
    const int j = (idx >> 10) & 15;
    const int k = idx & 1023;
    whh_s[idx] = (bf16)Whh[(size_t)k * 3072 + g * 1024 + col0 + j];
  }
  if (tid < 48) bias_s[tid] = bhh[(tid >> 4) * 1024 + col0 + (tid & 15)];
  if (tid < 128) hown[tid] = 0.f;
  __syncthreads();

  unsigned gen_target = 0;
  for (int t = 0; t < 512; ++t) {
    for (int idx = tid; idx < 768; idx += 256) acc[idx] = 0.f;
    __syncthreads();

    const bf16* hprev = hbuf + (size_t)(t & 1) * (16 * 1024);
    const int arow = lane & 15;
    v8f c[3] = {zero8(), zero8(), zero8()};
#pragma unroll
    for (int s = 0; s < 4; ++s) { // this wave's 128-wide K slice, 4 k-steps
      const int kc = (wave << 7) + (s << 5);
      const v16bf af = frag_a(hprev + arow * 1024 + kc, lane);
#pragma unroll
      for (int g = 0; g < 3; ++g) {
        const v16bf bfr = frag_b(&whh_s[((g << 4) + (lane & 15)) * 1024 + kc], lane);
        c[g] = wmma_bf16(af, bfr, c[g]);
      }
    }
    const int crow = (lane >> 4) << 3;
    const int ccol = lane & 15;
#pragma unroll
    for (int g = 0; g < 3; ++g)
#pragma unroll
      for (int r = 0; r < 8; ++r)
        atomicAdd(&acc[(g << 8) + ((crow + r) << 4) + ccol], c[g][r]); // ds_add_f32
    __syncthreads();

    if (tid < 128) {
      const int b = tid >> 4;
      const int j = tid & 15;
      const size_t row = (size_t)b * 512 + t;
      const int col = col0 + j;
      const float ghr = acc[(b << 4) + j] + bias_s[j];
      const float ghu = acc[256 + (b << 4) + j] + bias_s[16 + j];
      const float ghn = acc[512 + (b << 4) + j] + bias_s[32 + j];
      const float gir = GI[row * 3072 + col];
      const float giu = GI[row * 3072 + 1024 + col];
      const float gin = GI[row * 3072 + 2048 + col];
      const float rg = sigmoidf_(gir + ghr);
      const float ug = sigmoidf_(giu + ghu);
      const float ng = tanhf(gin + rg * ghn);
      const float hn = (1.f - ug) * ng + ug * hown[tid];
      hown[tid] = hn;
      const bf16 hb = (bf16)hn;
      hbuf[(size_t)((t + 1) & 1) * (16 * 1024) + b * 1024 + col] = hb;
      Hall[row * 1024 + col] = hb;
    }
    __syncthreads();
    if (tid == 0) { // device-wide sense barrier between timesteps
      __threadfence();
      ++gen_target;
      const unsigned prev = __hip_atomic_fetch_add(&bar[0], 1u, __ATOMIC_ACQ_REL,
                                                   __HIP_MEMORY_SCOPE_AGENT);
      if (prev == 63u) {
        __hip_atomic_store(&bar[0], 0u, __ATOMIC_RELAXED, __HIP_MEMORY_SCOPE_AGENT);
        __hip_atomic_store(&bar[1], gen_target, __ATOMIC_RELEASE,
                           __HIP_MEMORY_SCOPE_AGENT);
      } else {
        while (__hip_atomic_load(&bar[1], __ATOMIC_ACQUIRE,
                                 __HIP_MEMORY_SCOPE_AGENT) < gen_target)
          __builtin_amdgcn_s_sleep(1);
      }
    }
    __syncthreads();
  }
}

// ---------------------------------------------------------------------------
// K6: mu/sigma heads + reparameterized sample. K padded 100 -> 128.
// ---------------------------------------------------------------------------
__global__ __launch_bounds__(256) void heads_wmma(
    const bf16* __restrict__ Dp,   // (4096, 128), cols >= 100 are zero
    const float* __restrict__ Wmu, const float* __restrict__ bmu,
    const float* __restrict__ Wsg, const float* __restrict__ bsg,
    const float* __restrict__ noise, // (4096, 100), row = b*512 + t
    float* __restrict__ out) {       // (4096, 100)
  __shared__ __align__(16) bf16 WmS[112 * 128]; // [n][k], zero padded
  __shared__ __align__(16) bf16 WsS[112 * 128];
  const int tid = threadIdx.x;
  const int lane = tid & 31;
  const int wave = tid >> 5;
  for (int idx = tid; idx < 112 * 128; idx += 256) {
    const int n = idx >> 7;
    const int k = idx & 127;
    const bool ok = (n < 100) && (k < 100);
    WmS[idx] = ok ? (bf16)Wmu[k * 100 + n] : (bf16)0.f;
    WsS[idx] = ok ? (bf16)Wsg[k * 100 + n] : (bf16)0.f;
  }
  __syncthreads();
  const int m0 = ((int)blockIdx.x << 7) + (wave << 4);
  const int arow = lane & 15;
  v8f cm[7], cs[7];
#pragma unroll
  for (int j = 0; j < 7; ++j) { cm[j] = zero8(); cs[j] = zero8(); }
#pragma unroll
  for (int s = 0; s < 4; ++s) {
    const int kc = s << 5;
    const v16bf af = frag_a(Dp + (size_t)(m0 + arow) * 128 + kc, lane);
#pragma unroll
    for (int j = 0; j < 7; ++j) {
      const v16bf bm = frag_b(&WmS[((j << 4) + (lane & 15)) * 128 + kc], lane);
      cm[j] = wmma_bf16(af, bm, cm[j]);
      const v16bf bs = frag_b(&WsS[((j << 4) + (lane & 15)) * 128 + kc], lane);
      cs[j] = wmma_bf16(af, bs, cs[j]);
    }
  }
  const int crow = (lane >> 4) << 3;
  const int ccol = lane & 15;
#pragma unroll
  for (int j = 0; j < 7; ++j) {
    const int o = (j << 4) + ccol;
    if (o < 100) {
#pragma unroll
      for (int r = 0; r < 8; ++r) {
        const size_t row = (size_t)m0 + crow + r;
        const float mu = cm[j][r] + bmu[o];
        const float sg = cs[j][r] + bsg[o];
        const float sp = (sg > 20.f) ? sg : log1pf(__expf(sg));
        out[row * 100 + o] = mu + sp * noise[row * 100 + o];
      }
    }
  }
}

__global__ void init_ws_kernel(bf16* hbuf, unsigned* bar) {
  const int i = blockIdx.x * 256 + threadIdx.x;
  if (i < 2 * 16 * 1024) hbuf[i] = (bf16)0.f;
  if (i < 2) bar[i] = 0u;
}

__global__ void bias_comb_kernel(const float* __restrict__ blg,
                                 const float* __restrict__ Wih,
                                 const float* __restrict__ bih,
                                 float* __restrict__ bc) {
  const int j = blockIdx.x * 256 + threadIdx.x;
  if (j >= 3072) return;
  float s = bih[j];
  for (int k = 0; k < 4096; ++k) s += blg[k] * Wih[(size_t)k * 3072 + j];
  bc[j] = s;
}

extern "C" void kernel_launch(void* const* d_in, const int* in_sizes, int n_in,
                              void* d_out, int out_size, void* d_ws,
                              size_t ws_size, hipStream_t stream) {
  (void)in_sizes; (void)n_in; (void)out_size; (void)ws_size;
  const float* z = (const float*)d_in[0];
  const float* noise = (const float*)d_in[1];
  const float* Wlg = (const float*)d_in[2];
  const float* blg = (const float*)d_in[3];
  const float* Wih = (const float*)d_in[4];
  const float* bih = (const float*)d_in[5];
  const float* Whh = (const float*)d_in[6];
  const float* bhh = (const float*)d_in[7];
  const float* Wd = (const float*)d_in[8];
  const float* bd = (const float*)d_in[9];
  const float* Wmu = (const float*)d_in[10];
  const float* bmu = (const float*)d_in[11];
  const float* Wsg = (const float*)d_in[12];
  const float* bsg = (const float*)d_in[13];
  float* out = (float*)d_out;

  char* ws = (char*)d_ws;
  size_t off = 0;
  auto take = [&](size_t bytes) -> void* {
    void* p = ws + off;
    off = (off + bytes + 255) & ~(size_t)255;
    return p;
  };
  unsigned* bar = (unsigned*)take(2 * sizeof(unsigned));
  bf16* hbuf = (bf16*)take((size_t)2 * 16 * 1024 * sizeof(bf16));
  bf16* WcombT = (bf16*)take((size_t)4096 * 3072 * sizeof(bf16)); // [n=3072][k=4096]
  float* GI = (float*)take((size_t)4096 * 3072 * sizeof(float));
  bf16* Hall = (bf16*)take((size_t)4096 * 1024 * sizeof(bf16));
  bf16* Dpad = (bf16*)take((size_t)4096 * 128 * sizeof(bf16));
  float* bcomb = (float*)take((size_t)3072 * sizeof(float));

  init_ws_kernel<<<128, 256, 0, stream>>>(hbuf, bar);
  bias_comb_kernel<<<12, 256, 0, stream>>>(blg, Wih, bih, bcomb);
  // W_comb^T = bf16((W_lgssm @ W_ih))^T stored K-major  M=4096 N=3072 K=4096
  gemm_wmma<float, float, bf16, false, false, true><<<dim3(24, 32), 256, 0, stream>>>(
      Wlg, 4096, Wih, 3072, WcombT, 4096, nullptr, 3072, 4096);
  // GI = Z @ W_comb + b_comb (B staged via async DMA)   M=4096(b*512+t) N=3072 K=4096
  gemm_wmma<float, bf16, float, false, true, false><<<dim3(24, 32), 256, 0, stream>>>(
      z, 4096, WcombT, 4096, GI, 3072, bcomb, 3072, 4096);
  // 512-step GRU recurrence (persistent kernel, device barrier per step)
  const size_t smem =
      3 * 16 * 1024 * sizeof(bf16) + (3 * 256 + 128 + 48) * sizeof(float);
  gru_recur<<<64, 256, smem, stream>>>(Whh, bhh, GI, hbuf, Hall, bar);
  // Dpad = bf16(H_all @ W_dense + b_dense), zero-padded to 128 cols (async A stage)
  gemm_wmma<bf16, float, bf16, true, false, false><<<dim3(1, 32), 256, 0, stream>>>(
      Hall, 1024, Wd, 100, Dpad, 128, bd, 100, 1024);
  heads_wmma<<<32, 256, 0, stream>>>(Dpad, Wmu, bmu, Wsg, bsg, noise, out);
}
